// NeighbourSupport_27101243638142
// MI455X (gfx1250) — compile-verified, matrix operand order checked by reading, not run on hardware
//
#include <hip/hip_runtime.h>
#include <hip/hip_bf16.h>
#include <stdint.h>

typedef __attribute__((ext_vector_type(16))) _Float16 v16h;
typedef __attribute__((ext_vector_type(8)))  float    v8f;

#define HWSZ 9216      // 96*96
#define IMW  96
#define CCH  512
#define CKV  64
#define PTOT (8 * HWSZ)

// VGPR-pair -> K offset mapping for 16-bit WMMA A/B operands (16x16x32):
// lanes 0-15 hold K = {0..7, 16..23}, lanes 16-31 hold K = {8..15, 24..31}
__device__ __forceinline__ int k0_of(int j, int hi) {
    return ((j >> 2) << 4) + (hi << 3) + ((j & 3) << 1);
}

// ---------------------------------------------------------------------------
// Kernel 1: fused key1 (1x1 conv + bias + ReLU) and value (1x1 conv + bias).
// x: [B,512,96,96] f32 (channel-major). Outputs kpre/v: [P][64] f32 (pixel-major).
// One wave = 16 pixels x 64 out-channels of one branch; K-loop over 512.
// ---------------------------------------------------------------------------
__global__ __launch_bounds__(256) void kv1x1_kernel(
    const float* __restrict__ x,
    const float* __restrict__ key1_w, const float* __restrict__ key1_b,
    const float* __restrict__ value_w, const float* __restrict__ value_b,
    float* __restrict__ kpre, float* __restrict__ vout)
{
    const int lane = threadIdx.x & 31;
    const int wid  = threadIdx.x >> 5;
    const int l15  = lane & 15;
    const int hi   = lane >> 4;
    const int br   = wid & 1;                       // 0 = key branch, 1 = value branch
    const int p0   = blockIdx.x * 64 + (wid >> 1) * 16;
    const int bb   = p0 / HWSZ;
    const int hw0  = p0 % HWSZ;

    const float* wmat = br ? value_w : key1_w;
    const float* bias = br ? value_b : key1_b;
    float*       dst  = br ? vout    : kpre;

    v8f acc[4];
    const v8f vzero = {0.f,0.f,0.f,0.f,0.f,0.f,0.f,0.f};
    #pragma unroll
    for (int nt = 0; nt < 4; ++nt) acc[nt] = vzero;

    const float* xbase = x + (size_t)bb * CCH * HWSZ + hw0 + l15;

    for (int kk = 0; kk < CCH / 32; ++kk) {
        const int c0 = kk * 32;
        // A tile: 16 pixels x 32 channels, f32 -> f16
        v16h a;
        #pragma unroll
        for (int j = 0; j < 8; ++j) {
            const int k = c0 + k0_of(j, hi);
            const float a0 = xbase[(size_t)k * HWSZ];
            const float a1 = xbase[(size_t)(k + 1) * HWSZ];
            a[2*j]   = (_Float16)a0;
            a[2*j+1] = (_Float16)a1;
        }
        if (kk + 1 < CCH / 32)
            __builtin_prefetch(xbase + (size_t)(c0 + 32) * HWSZ, 0, 1);

        #pragma unroll
        for (int nt = 0; nt < 4; ++nt) {
            const int oc = nt * 16 + l15;
            const float* wrow = wmat + (size_t)oc * CCH + c0;
            v16h bmat;
            #pragma unroll
            for (int j = 0; j < 8; ++j) {
                const int k = k0_of(j, hi);
                const float2 wv = *(const float2*)(wrow + k);
                bmat[2*j]   = (_Float16)wv.x;
                bmat[2*j+1] = (_Float16)wv.y;
            }
            acc[nt] = __builtin_amdgcn_wmma_f32_16x16x32_f16(
                false, a, false, bmat, (short)0, acc[nt], false, false);
        }
    }

    // Epilogue: bias (+ReLU on key branch), store pixel-major [P][64]
    #pragma unroll
    for (int nt = 0; nt < 4; ++nt) {
        const int oc = nt * 16 + l15;
        const float bv = bias[oc];
        #pragma unroll
        for (int r = 0; r < 8; ++r) {
            const int m = r + 8 * hi;
            float val = acc[nt][r] + bv;
            if (!br) val = fmaxf(val, 0.0f);
            dst[(size_t)(p0 + m) * CKV + oc] = val;
        }
    }
}

// ---------------------------------------------------------------------------
// Kernel 2: per-pixel depthwise 3x3 + key3 + softmax + 9-tap weighted sum,
// then out-projection GEMM (64 -> 512) with WMMA, residual add, store.
// Block = 128 pixels (8 waves, 16 px each for phase 1); each wave then runs
// the GEMM for its 64-out-channel slice over all 8 pixel subtiles.
// ---------------------------------------------------------------------------
__global__ __launch_bounds__(256) void attn_out_kernel(
    const float* __restrict__ x,
    const float* __restrict__ kpre, const float* __restrict__ vin,
    const float* __restrict__ dw_w, const float* __restrict__ dw_b,
    const float* __restrict__ k3_w, const float* __restrict__ k3_b,
    const float* __restrict__ out_w, const float* __restrict__ out_b,
    float* __restrict__ out)
{
    __shared__ float        sDw[9 * 64];     // depthwise weights, [tap][ch]
    __shared__ float        sK3[9 * 64];     // key3 weights, [tap'][ch]
    __shared__ float        sDwB[64];
    __shared__ float        sK3B[9];
    __shared__ unsigned int sA[16 * 256];    // A staging: 8 subtiles x 2 ktiles x 32 lanes x 8 dw
    __shared__ float        sT[8 * 256];     // per-wave 16x16 transpose buffer

    const int tid  = threadIdx.x;
    const int lane = tid & 31;
    const int wid  = tid >> 5;
    const int l15  = lane & 15;
    const int hi   = lane >> 4;

    // stage small weights into LDS
    for (int i = tid; i < 9 * 64; i += 256) {
        const int t = i >> 6, c = i & 63;
        sDw[i] = dw_w[c * 9 + t];     // [CK,1,3,3] -> [tap][ch]
        sK3[i] = k3_w[i];             // [9,CK,1,1] already [tap'][ch]
    }
    if (tid < 64) sDwB[tid] = dw_b[tid];
    if (tid < 9)  sK3B[tid] = k3_b[tid];

    // register-resident B tiles for out-proj: wave covers out-chs [wid*64, wid*64+64)
    v16h Bt[4][2];
    #pragma unroll
    for (int nt = 0; nt < 4; ++nt) {
        const int n = wid * 64 + nt * 16 + l15;
        const float* wrow = out_w + (size_t)n * CKV;   // out_w: [512][64]
        #pragma unroll
        for (int t = 0; t < 2; ++t) {
            #pragma unroll
            for (int j = 0; j < 8; ++j) {
                const int k = t * 32 + k0_of(j, hi);
                const float2 wv = *(const float2*)(wrow + k);
                Bt[nt][t][2*j]   = (_Float16)wv.x;
                Bt[nt][t][2*j+1] = (_Float16)wv.y;
            }
        }
    }
    __syncthreads();

    // ---- phase 1: lane owns pixel p, channel half 'hi' (32 channels in A layout)
    const int p  = blockIdx.x * 128 + wid * 16 + l15;
    const int hw = p % HWSZ;
    const int py = hw / IMW, px = hw % IMW;

    float kdw[32];
    #pragma unroll
    for (int q = 0; q < 16; ++q) {
        const int kq = ((q >> 3) << 5) + k0_of(q & 7, hi);
        kdw[2*q]   = sDwB[kq];
        kdw[2*q+1] = sDwB[kq + 1];
    }
    #pragma unroll
    for (int t = 0; t < 9; ++t) {
        const int dy = t / 3 - 1, dx = t % 3 - 1;
        const int ny = py + dy, nx = px + dx;
        if (ny < 0 || ny >= IMW || nx < 0 || nx >= IMW) continue;
        const float* krow  = kpre + (size_t)(p + dy * IMW + dx) * CKV;
        const float* dwrow = sDw + t * 64;
        #pragma unroll
        for (int q = 0; q < 16; ++q) {
            const int kq = ((q >> 3) << 5) + k0_of(q & 7, hi);
            const float2 kv = *(const float2*)(krow + kq);
            const float2 wv = *(const float2*)(dwrow + kq);
            kdw[2*q]   += wv.x * kv.x;
            kdw[2*q+1] += wv.y * kv.y;
        }
    }

    // key3 projection (partial over this lane's 32 channels), combine halves, softmax
    float lt[9];
    #pragma unroll
    for (int tt = 0; tt < 9; ++tt) {
        const float* k3r = sK3 + tt * 64;
        float s = 0.f;
        #pragma unroll
        for (int q = 0; q < 16; ++q) {
            const int kq = ((q >> 3) << 5) + k0_of(q & 7, hi);
            const float2 wv = *(const float2*)(k3r + kq);
            s += wv.x * kdw[2*q] + wv.y * kdw[2*q+1];
        }
        s += __shfl_xor(s, 16, 32);        // combine channel halves (wave32)
        lt[tt] = s + sK3B[tt];
    }
    float mx = lt[0];
    #pragma unroll
    for (int tt = 1; tt < 9; ++tt) mx = fmaxf(mx, lt[tt]);
    float wsm[9], den = 0.f;
    #pragma unroll
    for (int tt = 0; tt < 9; ++tt) { wsm[tt] = __expf(lt[tt] - mx); den += wsm[tt]; }
    const float rden = 1.f / den;
    #pragma unroll
    for (int tt = 0; tt < 9; ++tt) wsm[tt] *= rden;

    // 9-tap weighted sum of value neighborhoods
    float yac[32];
    #pragma unroll
    for (int i = 0; i < 32; ++i) yac[i] = 0.f;
    #pragma unroll
    for (int t = 0; t < 9; ++t) {
        const int dy = t / 3 - 1, dx = t % 3 - 1;
        const int ny = py + dy, nx = px + dx;
        if (ny < 0 || ny >= IMW || nx < 0 || nx >= IMW) continue;
        const float* vrow = vin + (size_t)(p + dy * IMW + dx) * CKV;
        const float wt = wsm[t];
        #pragma unroll
        for (int q = 0; q < 16; ++q) {
            const int kq = ((q >> 3) << 5) + k0_of(q & 7, hi);
            const float2 vv = *(const float2*)(vrow + kq);
            yac[2*q]   += wt * vv.x;
            yac[2*q+1] += wt * vv.y;
        }
    }

    // pack y -> f16 A tiles in LDS (already in WMMA A register layout)
    #pragma unroll
    for (int t = 0; t < 2; ++t) {
        #pragma unroll
        for (int j = 0; j < 8; ++j) {
            union { _Float16 h[2]; unsigned int u; } pk;
            const int q = t * 8 + j;
            pk.h[0] = (_Float16)yac[2*q];
            pk.h[1] = (_Float16)yac[2*q+1];
            sA[(wid * 2 + t) * 256 + lane * 8 + j] = pk.u;
        }
    }
    __syncthreads();

    // ---- phase 2: out-projection GEMM + residual, coalesced channel-major store
    const int blk0 = blockIdx.x * 128;
    #pragma unroll 1
    for (int s = 0; s < 8; ++s) {
        union { v16h v; unsigned int u[8]; } A0, A1;
        #pragma unroll
        for (int j = 0; j < 8; ++j) {
            A0.u[j] = sA[(s * 2 + 0) * 256 + lane * 8 + j];
            A1.u[j] = sA[(s * 2 + 1) * 256 + lane * 8 + j];
        }
        const int p0  = blk0 + s * 16;
        const int bb  = p0 / HWSZ;
        const int hw0 = p0 % HWSZ;
        float* T = sT + wid * 256;
        #pragma unroll 1
        for (int nt = 0; nt < 4; ++nt) {
            const int n0 = wid * 64 + nt * 16;
            v8f acc = {0.f,0.f,0.f,0.f,0.f,0.f,0.f,0.f};
            acc = __builtin_amdgcn_wmma_f32_16x16x32_f16(
                false, A0.v, false, Bt[nt][0], (short)0, acc, false, false);
            acc = __builtin_amdgcn_wmma_f32_16x16x32_f16(
                false, A1.v, false, Bt[nt][1], (short)0, acc, false, false);

            // transpose 16x16 tile through wave-private LDS (lanes=N -> lanes=M)
            #pragma unroll
            for (int r = 0; r < 8; ++r)
                T[l15 * 16 + r + 8 * hi] = acc[r];
            __asm__ volatile("s_wait_dscnt 0" ::: "memory");
            #pragma unroll
            for (int rr = 0; rr < 8; ++rr) {
                const int c = rr * 2 + hi;
                const float val = T[c * 16 + l15];
                const size_t g = (size_t)(bb * CCH + n0 + c) * HWSZ + hw0 + l15;
                out[g] = val + out_b[n0 + c] + x[g];
            }
            __asm__ volatile("s_wait_dscnt 0" ::: "memory");
        }
    }
}

extern "C" void kernel_launch(void* const* d_in, const int* in_sizes, int n_in,
                              void* d_out, int out_size, void* d_ws, size_t ws_size,
                              hipStream_t stream)
{
    const float* x        = (const float*)d_in[0];
    const float* key1_w   = (const float*)d_in[1];
    const float* key1_b   = (const float*)d_in[2];
    const float* key_dw_w = (const float*)d_in[3];
    const float* key_dw_b = (const float*)d_in[4];
    const float* key3_w   = (const float*)d_in[5];
    const float* key3_b   = (const float*)d_in[6];
    const float* value_w  = (const float*)d_in[7];
    const float* value_b  = (const float*)d_in[8];
    const float* out_w    = (const float*)d_in[9];
    const float* out_b    = (const float*)d_in[10];
    float* out = (float*)d_out;

    float* kpre = (float*)d_ws;                         // [P][64] f32
    float* vbuf = kpre + (size_t)PTOT * CKV;            // [P][64] f32

    kv1x1_kernel<<<PTOT / 64, 256, 0, stream>>>(
        x, key1_w, key1_b, value_w, value_b, kpre, vbuf);
    attn_out_kernel<<<PTOT / 128, 256, 0, stream>>>(
        x, kpre, vbuf, key_dw_w, key_dw_b, key3_w, key3_b, out_w, out_b, out);
}
